// ADI_16389595202112
// MI455X (gfx1250) — compile-verified, matrix operand order checked by reading, not run on hardware
//
#include <hip/hip_runtime.h>
#include <hip/hip_bf16.h>
#include <math.h>

// ---------------------------------------------------------------------------
// MI455X (gfx1250) implementation of the ADI two-tower model.
// GEMMs use V_WMMA_F32_16X16X32_F16 (f16 in, f32 accumulate). All elementwise
// epilogues (bias, eval-BN, ReLU, gate-weighted expert mixing) are fused into
// the WMMA kernel. Per-wave tile is 32x(NT*16): each B (weight) fragment is
// reused by two WMMAs, cutting b128 loads per WMMA from 2.5 to 1.5.
// Working set (~70MB) is L2-resident (192MB L2).
// ---------------------------------------------------------------------------

typedef __attribute__((ext_vector_type(16))) _Float16 v16h;
typedef __attribute__((ext_vector_type(8)))  _Float16 v8h;
typedef __attribute__((ext_vector_type(8)))  float    v8f;

#define BN_EPS 1e-5f
constexpr int Bn = 16384;
constexpr int L  = 128;
constexpr int H0 = 256, H1 = 128, H2 = 64, H3 = 32;
constexpr int E  = 8, T = 5;

// ---------------- gather embeddings (f32 + f16 copies) ----------------------
__global__ void gather_kernel(const int* __restrict__ idx,
                              const float* __restrict__ table,
                              float* __restrict__ outF,
                              _Float16* __restrict__ outH)
{
    int b = blockIdx.x;          // one block per sample
    int c = threadIdx.x;         // blockDim.x == L == 128
    float v = table[(long long)idx[b] * L + c];
    outF[b * L + c] = v;
    outH[b * L + c] = (_Float16)v;
}

// -------- copy a domain-selected f32 slice (bias / BN params) ---------------
__global__ void select_f32(const float* __restrict__ in, float* __restrict__ out,
                           int n, const int* __restrict__ dptr, int dstride)
{
    int i = blockIdx.x * blockDim.x + threadIdx.x;
    if (i >= n) return;
    int d = dptr ? *dptr : 0;
    out[i] = in[(long long)d * dstride + i];
}

// -------- weight prep: f32 [K,N] (domain slice) -> f16 [N,K] (transposed) ---
// Transposed layout lets B-fragment loads be two contiguous 16B loads/lane.
__global__ void transpose_f16(const float* __restrict__ in, _Float16* __restrict__ out,
                              int K, int N, const int* __restrict__ dptr, int dstride)
{
    int i = blockIdx.x * blockDim.x + threadIdx.x;
    if (i >= K * N) return;
    int d = dptr ? *dptr : 0;
    int k = i / N, n = i - k * N;
    out[(long long)n * K + k] = (_Float16)in[(long long)d * dstride + i];
}

// ---------------- WMMA GEMM with fused epilogue -----------------------------
// A [M,K] f16 row-major, Wt [N,K] f16 row-major (pre-transposed weights).
// Each block: 8 waves; each wave computes a 32x(NT*16) tile, K stepped by 32.
// Epilogue: +bias, eval-BN (optional), ReLU, optional gate scale & accumulate.
template <int NT>
__global__ void __launch_bounds__(256) gemm_wmma(
    const _Float16* __restrict__ A,    // [M,K]
    const _Float16* __restrict__ Wt,   // [N,K]
    const float* __restrict__ bias,    // [N]
    const float* __restrict__ bn,      // [4,N] (gamma,beta,mean,var) or null
    float*     __restrict__ outF,      // [M,N] or null
    _Float16*  __restrict__ outH,      // [M,N] or null
    const float* __restrict__ gate,    // [M,E] or null (expert-mix fusion)
    int gate_idx, int accumulate,
    int M, int N, int K)
{
    const int lane  = threadIdx.x & 31;
    const int wave  = threadIdx.x >> 5;
    const int mbase = (blockIdx.y * 8 + wave) * 32;   // 32 rows per wave
    const int nbase = blockIdx.x * (NT * 16);
    const int hlane = lane & 15;
    const int kb    = (lane >> 4) * 8;      // K-phase: lanes 0-15 -> 0, 16-31 -> 8

    v8f acc[2][NT] = {};

    const _Float16* ap0 = A + (long long)(mbase + hlane) * K + kb;
    const _Float16* ap1 = ap0 + (long long)16 * K;
    for (int kk = 0; kk < K; kk += 32) {
        // A fragments for the two 16-row subtiles (7.12.2 layout:
        // per lane K = kb+0..7 and kb+16..23 within the 32-wide K slice)
        union { v16h v; v8h h[2]; } fa0, fa1;
        fa0.h[0] = *(const v8h*)(ap0 + kk);
        fa0.h[1] = *(const v8h*)(ap0 + kk + 16);
        fa1.h[0] = *(const v8h*)(ap1 + kk);
        fa1.h[1] = *(const v8h*)(ap1 + kk + 16);
        // speculative prefetch of next A K-tile (global_prefetch_b8)
        __builtin_prefetch(ap0 + kk + 32, 0, 1);
        __builtin_prefetch(ap1 + kk + 32, 0, 1);
#pragma unroll
        for (int t = 0; t < NT; ++t) {
            const _Float16* bp = Wt + (long long)(nbase + t * 16 + hlane) * K + kk + kb;
            union { v16h v; v8h h[2]; } fb;
            fb.h[0] = *(const v8h*)(bp);
            fb.h[1] = *(const v8h*)(bp + 16);
            // B fragment reused by both M subtiles
            acc[0][t] = __builtin_amdgcn_wmma_f32_16x16x32_f16(
                false, fa0.v, false, fb.v, (short)0, acc[0][t], false, false);
            acc[1][t] = __builtin_amdgcn_wmma_f32_16x16x32_f16(
                false, fa1.v, false, fb.v, (short)0, acc[1][t], false, false);
        }
    }

    // C/D layout: lane -> column (n = hlane), VGPR r -> row (m = r, +8 hi lanes)
#pragma unroll
    for (int t = 0; t < NT; ++t) {
        const int n = nbase + t * 16 + hlane;
        const float bsv = bias[n];
        float bt = 0.f, mn = 0.f, scale = 1.f;
        if (bn) {
            mn    = bn[2 * N + n];
            bt    = bn[N + n];
            scale = bn[n] * rsqrtf(bn[3 * N + n] + BN_EPS);
        }
#pragma unroll
        for (int s = 0; s < 2; ++s) {
            const int mrow0 = mbase + s * 16 + ((lane >> 4) << 3);
#pragma unroll
            for (int r = 0; r < 8; ++r) {
                int m = mrow0 + r;
                float v = acc[s][t][r] + bsv;
                if (bn) v = (v - mn) * scale + bt;
                v = v > 0.f ? v : 0.f;
                if (gate) v *= gate[(long long)m * E + gate_idx];
                long long o = (long long)m * N + n;
                if (outF) outF[o] = accumulate ? outF[o] + v : v;
                if (outH) outH[o] = (_Float16)v;
            }
        }
    }
}

// ---------------- gate heads: softmax(task-d gate) + sigmoid(ADI) -----------
__global__ void gate_adi_kernel(const float* __restrict__ embF,
                                const float* __restrict__ dom_base,  // [2T,L]
                                int side_off,
                                const float* __restrict__ gate_w,    // [T,L,E]
                                const float* __restrict__ gate_b,    // [T,E]
                                const float* __restrict__ adi_w,     // [L,2]
                                const float* __restrict__ adi_b,     // [2]
                                const int* __restrict__ dptr,
                                float* __restrict__ g,               // [B,E]
                                float* __restrict__ ga)              // [B,2]
{
    int gid  = blockIdx.x * blockDim.x + threadIdx.x;
    int b    = gid >> 5;                  // one wave32 per sample
    int lane = gid & 31;
    if (b >= Bn) return;
    int d = *dptr;
    const float* gw  = gate_w   + (long long)d * L * E;
    const float* dom = dom_base + (long long)(d + side_off) * L;

    float ae[E] = {};
    float a2[2] = {};
    for (int l = lane; l < L; l += 32) {
        float x  = embF[(long long)b * L + l];
        float xd = x + dom[l];
#pragma unroll
        for (int e = 0; e < E; ++e) ae[e] += x * gw[l * E + e];
        a2[0] += xd * adi_w[l * 2 + 0];
        a2[1] += xd * adi_w[l * 2 + 1];
    }
#pragma unroll
    for (int off = 16; off > 0; off >>= 1) {
#pragma unroll
        for (int e = 0; e < E; ++e) ae[e] += __shfl_down(ae[e], off);
        a2[0] += __shfl_down(a2[0], off);
        a2[1] += __shfl_down(a2[1], off);
    }
    if (lane == 0) {
        float lg[E], mx = -1e30f;
#pragma unroll
        for (int e = 0; e < E; ++e) { lg[e] = ae[e] + gate_b[d * E + e]; mx = fmaxf(mx, lg[e]); }
        float s = 0.f;
#pragma unroll
        for (int e = 0; e < E; ++e) { lg[e] = __expf(lg[e] - mx); s += lg[e]; }
        float inv = 1.f / s;
#pragma unroll
        for (int e = 0; e < E; ++e) g[(long long)b * E + e] = lg[e] * inv;
        ga[b * 2 + 0] = 1.f / (1.f + __expf(-(a2[0] + adi_b[0])));
        ga[b * 2 + 1] = 1.f / (1.f + __expf(-(a2[1] + adi_b[1])));
    }
}

// ---------------- ADI combine: [spec, spec*share, share] -> f16 [B,384] -----
__global__ void combine_kernel(const float* __restrict__ expertF,
                               const float* __restrict__ weightedF,
                               const float* __restrict__ ga,
                               _Float16* __restrict__ adiH)
{
    int i = blockIdx.x * blockDim.x + threadIdx.x;
    if (i >= Bn * H1) return;
    int b = i >> 7;       // H1 == 128
    int h = i & 127;
    float spec  = expertF[i]   * ga[b * 2 + 1];
    float share = weightedF[i] * ga[b * 2 + 0];
    long long o = (long long)b * (3 * H1);
    adiH[o + h]          = (_Float16)spec;
    adiH[o + H1 + h]     = (_Float16)(spec * share);
    adiH[o + 2 * H1 + h] = (_Float16)share;
}

// ---------------- final per-row dot of the two towers -----------------------
__global__ void dot_kernel(const float* __restrict__ u, const float* __restrict__ v,
                           float* __restrict__ out)
{
    int gid  = blockIdx.x * blockDim.x + threadIdx.x;
    int b    = gid >> 5;                 // one wave32 per sample, 32 lanes == H3
    int lane = gid & 31;
    if (b >= Bn) return;
    float p = u[(long long)b * H3 + lane] * v[(long long)b * H3 + lane];
#pragma unroll
    for (int off = 16; off > 0; off >>= 1) p += __shfl_down(p, off);
    if (lane == 0) out[b] = p;
}

// ---------------------------------------------------------------------------
extern "C" void kernel_launch(void* const* d_in, const int* in_sizes, int n_in,
                              void* d_out, int out_size, void* d_ws, size_t ws_size,
                              hipStream_t stream)
{
    const int* user_idx = (const int*)d_in[0];
    const int* item_idx = (const int*)d_in[1];
    const int* dptr     = (const int*)d_in[2];   // domain index lives on device
    const float* emb_user = (const float*)d_in[45];
    const float* emb_item = (const float*)d_in[46];
    const float* dom_embs = (const float*)d_in[47];

    // ---- workspace carve-up (~70 MB total; buffers reused across sides) ----
    char* p = (char*)d_ws;
    auto carve = [&](size_t bytes) -> char* {
        char* r = p; p += (bytes + 255) & ~(size_t)255; return r;
    };
    float*    embF = (float*)   carve((size_t)Bn * L * 4);
    _Float16* embH = (_Float16*)carve((size_t)Bn * L * 2);
    _Float16* hH   = (_Float16*)carve((size_t)Bn * H0 * 2);
    _Float16* s1H  = (_Float16*)carve((size_t)Bn * H0 * 2);
    float*    expF = (float*)   carve((size_t)Bn * H1 * 4);
    float*    wgtF = (float*)   carve((size_t)Bn * H1 * 4);
    float*    gbuf = (float*)   carve((size_t)Bn * E * 4);
    float*    gabuf= (float*)   carve((size_t)Bn * 2 * 4);
    _Float16* adiH = (_Float16*)carve((size_t)Bn * 3 * H1 * 2);
    _Float16* h3H  = (_Float16*)carve((size_t)Bn * 3 * H2 * 2);
    float*    uOut = (float*)   carve((size_t)Bn * H3 * 4);
    float*    iOut = (float*)   carve((size_t)Bn * H3 * 4);
    _Float16* wt1  = (_Float16*)carve((size_t)H0 * L * 2);        // [256,128]
    _Float16* wt2  = (_Float16*)carve((size_t)H1 * H0 * 2);       // [128,256]
    _Float16* wt3  = (_Float16*)carve((size_t)3*H2 * 3*H1 * 2);   // [192,384]
    _Float16* wt4  = (_Float16*)carve((size_t)H3 * 3*H2 * 2);     // [32,192]
    float*    biasBuf = (float*)carve(H0 * 4);
    float*    bnBuf   = (float*)carve(4 * H0 * 4);

    auto run_side = [&](const int* idx, const float* tab, int side_off,
                        const float* sfc1_w, const float* sfc1_b, const float* dsbn1,
                        const float* sfc2_w, const float* sfc2_b, const float* dsbn2,
                        const float* m1_w,  const float* m1_b,  const float* bn1,
                        const float* m2_w,  const float* m2_b,  const float* bn2,
                        const float* gate_w, const float* gate_b,
                        const float* adi_w,  const float* adi_b,
                        const float* m3_w,  const float* m3_b,  const float* bn3,
                        const float* m4_w,  const float* m4_b,
                        float* sideOut)
    {
        gather_kernel<<<Bn, L, 0, stream>>>(idx, tab, embF, embH);
        gate_adi_kernel<<<(Bn * 32 + 255) / 256, 256, 0, stream>>>(
            embF, dom_embs, side_off, gate_w, gate_b, adi_w, adi_b, dptr, gbuf, gabuf);

        // task tower layer 1: [B,128] x [128,256] + bn1 + relu -> hH
        transpose_f16<<<(L * H0 + 255) / 256, 256, 0, stream>>>(m1_w, wt1, L, H0, dptr, L * H0);
        select_f32<<<1, 256, 0, stream>>>(m1_b, biasBuf, H0, dptr, H0);
        select_f32<<<4, 256, 0, stream>>>(bn1, bnBuf, 4 * H0, dptr, 4 * H0);
        gemm_wmma<4><<<dim3(H0 / 64, Bn / 256), 256, 0, stream>>>(
            embH, wt1, biasBuf, bnBuf, nullptr, hH, nullptr, 0, 0, Bn, H0, L);

        // task tower layer 2: [B,256] x [256,128] + bn2 + relu -> expF (f32)
        transpose_f16<<<(H0 * H1 + 255) / 256, 256, 0, stream>>>(m2_w, wt2, H0, H1, dptr, H0 * H1);
        select_f32<<<1, 128, 0, stream>>>(m2_b, biasBuf, H1, dptr, H1);
        select_f32<<<2, 256, 0, stream>>>(bn2, bnBuf, 4 * H1, dptr, 4 * H1);
        gemm_wmma<4><<<dim3(H1 / 64, Bn / 256), 256, 0, stream>>>(
            hH, wt2, biasBuf, bnBuf, expF, nullptr, nullptr, 0, 0, Bn, H1, H0);

        // shared experts (sequential; expert mixing fused into sfc2 epilogue)
        for (int e = 0; e < E; ++e) {
            transpose_f16<<<(L * H0 + 255) / 256, 256, 0, stream>>>(
                sfc1_w + (size_t)e * L * H0, wt1, L, H0, nullptr, 0);
            select_f32<<<4, 256, 0, stream>>>(dsbn1 + (size_t)e * 4 * H0, bnBuf,
                                              4 * H0, dptr, E * 4 * H0);
            gemm_wmma<4><<<dim3(H0 / 64, Bn / 256), 256, 0, stream>>>(
                embH, wt1, sfc1_b + (size_t)e * H0, bnBuf,
                nullptr, s1H, nullptr, 0, 0, Bn, H0, L);

            transpose_f16<<<(H0 * H1 + 255) / 256, 256, 0, stream>>>(
                sfc2_w + (size_t)e * H0 * H1, wt2, H0, H1, nullptr, 0);
            select_f32<<<2, 256, 0, stream>>>(dsbn2 + (size_t)e * 4 * H1, bnBuf,
                                              4 * H1, dptr, E * 4 * H1);
            // wgtF (+)= relu(bn(s1H @ sfc2)) * g[:,e]   -- einsum fusion
            gemm_wmma<4><<<dim3(H1 / 64, Bn / 256), 256, 0, stream>>>(
                s1H, wt2, sfc2_b + (size_t)e * H1, bnBuf,
                wgtF, nullptr, gbuf, e, (e > 0) ? 1 : 0, Bn, H1, H0);
        }

        combine_kernel<<<(Bn * H1 + 255) / 256, 256, 0, stream>>>(expF, wgtF, gabuf, adiH);

        // m3: [B,384] x [384,192] + bn3 + relu -> h3H
        transpose_f16<<<(3 * H1 * 3 * H2 + 255) / 256, 256, 0, stream>>>(
            m3_w, wt3, 3 * H1, 3 * H2, dptr, 3 * H1 * 3 * H2);
        select_f32<<<1, 192, 0, stream>>>(m3_b, biasBuf, 3 * H2, dptr, 3 * H2);
        select_f32<<<3, 256, 0, stream>>>(bn3, bnBuf, 4 * 3 * H2, dptr, 4 * 3 * H2);
        gemm_wmma<4><<<dim3(3 * H2 / 64, Bn / 256), 256, 0, stream>>>(
            adiH, wt3, biasBuf, bnBuf, nullptr, h3H, nullptr, 0, 0, Bn, 3 * H2, 3 * H1);

        // m4: [B,192] x [192,32] + bias + relu -> sideOut (f32)
        transpose_f16<<<(3 * H2 * H3 + 255) / 256, 256, 0, stream>>>(
            m4_w, wt4, 3 * H2, H3, dptr, 3 * H2 * H3);
        select_f32<<<1, 32, 0, stream>>>(m4_b, biasBuf, H3, dptr, H3);
        gemm_wmma<2><<<dim3(1, Bn / 256), 256, 0, stream>>>(
            h3H, wt4, biasBuf, nullptr, sideOut, nullptr, nullptr, 0, 0, Bn, H3, 3 * H2);
    };

    auto f = [&](int i) { return (const float*)d_in[i]; };
    const int ub = 3, ib = 24;
    run_side(user_idx, emb_user, 0,
             f(ub + 0),  f(ub + 1),  f(ub + 2),  f(ub + 3),  f(ub + 4),  f(ub + 5),
             f(ub + 6),  f(ub + 7),  f(ub + 8),  f(ub + 9),  f(ub + 10), f(ub + 11),
             f(ub + 12), f(ub + 13), f(ub + 14), f(ub + 15),
             f(ub + 16), f(ub + 17), f(ub + 18), f(ub + 19), f(ub + 20), uOut);
    run_side(item_idx, emb_item, T,
             f(ib + 0),  f(ib + 1),  f(ib + 2),  f(ib + 3),  f(ib + 4),  f(ib + 5),
             f(ib + 6),  f(ib + 7),  f(ib + 8),  f(ib + 9),  f(ib + 10), f(ib + 11),
             f(ib + 12), f(ib + 13), f(ib + 14), f(ib + 15),
             f(ib + 16), f(ib + 17), f(ib + 18), f(ib + 19), f(ib + 20), iOut);

    dot_kernel<<<(Bn * 32 + 255) / 256, 256, 0, stream>>>(uOut, iOut, (float*)d_out);
}
